// SimpleFlashBigBirdAttention_90537910600186
// MI455X (gfx1250) — matrix operand order repaired and checked by reading, not compiled, not run on hardware
//
#include <hip/hip_runtime.h>

// ---------------------------------------------------------------------------
// MHA forward for MI455X (gfx1250): bf16 WMMA everywhere, fp32 accumulate.
// Data staging uses GLOBAL_LOAD_ASYNC_TO_LDS_B128 (ASYNCcnt) when the
// toolchain exposes the builtin; otherwise falls back to b128 load/ds_store.
// ---------------------------------------------------------------------------

typedef unsigned short ushort_t;
typedef __attribute__((ext_vector_type(16))) __bf16 v16bf;
typedef __attribute__((ext_vector_type(8)))  float  v8f;
typedef __attribute__((ext_vector_type(4)))  int    i32x4;

#define DIM     768
#define HEADS   12
#define HD      64
#define BATCH   8
#define SEQ     1024
#define M_TOT   (BATCH * SEQ)     // 8192 tokens
#define C3      (3 * DIM)         // 2304
#define BHTOT   (BATCH * HEADS)   // 96
#define SCALE_F 0.03608439182435161f   // 768^-0.5 (full dim, per reference)
#define L2E     1.4426950408889634f

// ---- async global->LDS staging (CDNA5), guarded for toolchain support -----
#if defined(__has_builtin)
#  if __has_builtin(__builtin_amdgcn_global_load_async_to_lds_b128) && \
      __has_builtin(__builtin_amdgcn_s_wait_asynccnt)
#    define ASYNC_LDS 1
#  endif
#endif
#ifndef ASYNC_LDS
#  define ASYNC_LDS 0
#endif

#if ASYNC_LDS
#define AS_GLOBAL __attribute__((address_space(1)))
#define AS_LDS    __attribute__((address_space(3)))
__device__ __forceinline__ void stage16(void* lds, const void* g) {
    __builtin_amdgcn_global_load_async_to_lds_b128(
        (AS_GLOBAL i32x4*)(g), (AS_LDS i32x4*)(lds), 0, 0);
}
__device__ __forceinline__ void stage_wait() {
    __builtin_amdgcn_s_wait_asynccnt(0);
}
#else
__device__ __forceinline__ void stage16(void* lds, const void* g) {
    *(uint4*)lds = *(const uint4*)g;
}
__device__ __forceinline__ void stage_wait() {}
#endif

__device__ __forceinline__ ushort_t f2bf(float f) {
    union { float f; unsigned u; } c; c.f = f;
    unsigned r = c.u + 0x7FFFu + ((c.u >> 16) & 1u);   // round-to-nearest-even
    return (ushort_t)(r >> 16);
}

__device__ __forceinline__ v8f v8f_zero() {
    v8f v;
    #pragma unroll
    for (int i = 0; i < 8; ++i) v[i] = 0.0f;
    return v;
}

union Frag16 { uint4 u[2]; v16bf v; };   // 32B: one bf16 A/B WMMA fragment per lane

// ---------------------------------------------------------------------------
// Pack kernels: fp32 -> bf16 (optionally transposed so GEMM B-tiles are [n][k])
// ---------------------------------------------------------------------------
__global__ void pack_bf16(const float* __restrict__ in, ushort_t* __restrict__ out, int n) {
    for (int i = blockIdx.x * blockDim.x + threadIdx.x; i < n; i += gridDim.x * blockDim.x)
        out[i] = f2bf(in[i]);
}

__global__ void pack_transpose_bf16(const float* __restrict__ in, ushort_t* __restrict__ out,
                                    int K, int N) {   // in[K][N] -> out[N][K]
    int total = K * N;
    for (int i = blockIdx.x * blockDim.x + threadIdx.x; i < total; i += gridDim.x * blockDim.x) {
        int n = i / K, k = i - n * K;
        out[i] = f2bf(in[k * N + n]);
    }
}

// ---------------------------------------------------------------------------
// GEMM1: qkv = x @ W_qkv + b, scatter-store into Q[bh][n][d], K[bh][n][d],
//        VT[bh][d][n] as bf16.  M=8192, N=2304, K=768.  128x128x32 tiles.
// ---------------------------------------------------------------------------
__global__ __launch_bounds__(256)
void gemm_qkv(const ushort_t* __restrict__ Ab,    // [8192][768] bf16
              const ushort_t* __restrict__ Bt,    // [2304][768] bf16 (W^T)
              const float*    __restrict__ bias,  // [2304]
              ushort_t* __restrict__ Qo,
              ushort_t* __restrict__ Ko,
              ushort_t* __restrict__ VTo)
{
    __shared__ __align__(16) ushort_t As[2][128][32];
    __shared__ __align__(16) ushort_t Bs[2][128][32];

    const int mBase = blockIdx.x * 128;
    const int nBase = blockIdx.y * 128;
    const int t    = threadIdx.x;
    const int wave = t >> 5;
    const int lane = t & 31;
    const int half = lane >> 4;
    const int l16  = lane & 15;
    const int wm   = (wave & 3) * 32;   // 4 wave-rows
    const int wn   = (wave >> 2) * 64;  // 2 wave-cols

    v8f acc[2][4];
    #pragma unroll
    for (int i = 0; i < 2; ++i)
        #pragma unroll
        for (int j = 0; j < 4; ++j) acc[i][j] = v8f_zero();

    auto loadTiles = [&](int buf, int k0) {
        #pragma unroll
        for (int idx = t; idx < 512; idx += 256) {
            int row = idx >> 2, c8 = idx & 3;
            stage16(&As[buf][row][c8 * 8], &Ab[(size_t)(mBase + row) * DIM + k0 + c8 * 8]);
            stage16(&Bs[buf][row][c8 * 8], &Bt[(size_t)(nBase + row) * DIM + k0 + c8 * 8]);
        }
    };

    loadTiles(0, 0);
    int buf = 0;
    for (int k0 = 0; k0 < DIM; k0 += 32) {
        stage_wait();          // my async copies into `buf` (issued last iter) done
        __syncthreads();       // everyone's copies visible; prev compute done
        if (k0 + 32 < DIM) loadTiles(buf ^ 1, k0 + 32);
        if (k0 + 64 < DIM) {  // prefetch tile-after-next into GL2 (global_prefetch_b8)
            __builtin_prefetch(&Ab[(size_t)(mBase + (t & 127)) * DIM + k0 + 64]);
            __builtin_prefetch(&Bt[(size_t)(nBase + (t & 127)) * DIM + k0 + 64]);
        }
        Frag16 a[2], b[4];
        #pragma unroll
        for (int i = 0; i < 2; ++i) {
            int m = wm + i * 16 + l16;
            a[i].u[0] = *(const uint4*)&As[buf][m][half * 8];
            a[i].u[1] = *(const uint4*)&As[buf][m][16 + half * 8];
        }
        #pragma unroll
        for (int j = 0; j < 4; ++j) {
            int n = wn + j * 16 + l16;
            b[j].u[0] = *(const uint4*)&Bs[buf][n][half * 8];
            b[j].u[1] = *(const uint4*)&Bs[buf][n][16 + half * 8];
        }
        #pragma unroll
        for (int i = 0; i < 2; ++i)
            #pragma unroll
            for (int j = 0; j < 4; ++j)
                acc[i][j] = __builtin_amdgcn_wmma_f32_16x16x32_bf16(
                    false, a[i].v, false, b[j].v, (short)0, acc[i][j], false, false);
        buf ^= 1;
    }

    // Epilogue: bias + scatter into Q / K / V^T (bf16)
    #pragma unroll
    for (int j = 0; j < 4; ++j) {
        int c = nBase + wn + j * 16 + l16;
        float bv = bias[c];
        int which = c / DIM;
        int c7 = c - which * DIM;
        int h = c7 >> 6, d = c7 & 63;
        #pragma unroll
        for (int i = 0; i < 2; ++i) {
            #pragma unroll
            for (int r = 0; r < 8; ++r) {
                int m  = mBase + wm + i * 16 + half * 8 + r;
                int bb = m >> 10, tok = m & 1023;
                size_t bh = (size_t)(bb * HEADS + h);
                ushort_t val = f2bf(acc[i][j][r] + bv);
                if (which == 0)      Qo [bh * (SEQ * HD) + tok * HD + d] = val;
                else if (which == 1) Ko [bh * (SEQ * HD) + tok * HD + d] = val;
                else                 VTo[bh * (SEQ * HD) + d * SEQ + tok] = val;
            }
        }
    }
}

// ---------------------------------------------------------------------------
// Flash attention: 4 waves/block, each wave owns a 16-row Q tile; key blocks
// of 32 double-buffered through LDS; online softmax in C-fragment layout.
// ---------------------------------------------------------------------------
__global__ __launch_bounds__(128)
void flash_attn(const ushort_t* __restrict__ Qc,
                const ushort_t* __restrict__ Kc,
                const ushort_t* __restrict__ VT,
                ushort_t* __restrict__ ctx)     // [8192][768] bf16
{
    __shared__ __align__(16) ushort_t Ks[2][32][64];   // [key][d]
    __shared__ __align__(16) ushort_t Vs[2][64][32];   // [d][key]  (from V^T)
    __shared__ __align__(16) ushort_t Ps[4][16][32];   // wave-private P tiles

    const int bh    = blockIdx.x;          // 0..95
    const int qBase = blockIdx.y * 64;
    const int b = bh / HEADS;
    const int h = bh - b * HEADS;
    const int t    = threadIdx.x;
    const int wave = t >> 5;
    const int lane = t & 31;
    const int half = lane >> 4;
    const int l16  = lane & 15;
    const size_t boff = (size_t)bh * (SEQ * HD);

    // Q fragments: 16 rows x 64 d -> two 16x32 bf16 A-fragments
    Frag16 aq[2];
    {
        int m = qBase + wave * 16 + l16;
        const ushort_t* qr = Qc + boff + (size_t)m * HD;
        #pragma unroll
        for (int c = 0; c < 2; ++c) {
            aq[c].u[0] = *(const uint4*)&qr[c * 32 + half * 8];
            aq[c].u[1] = *(const uint4*)&qr[c * 32 + 16 + half * 8];
        }
    }

    v8f accO[4];
    #pragma unroll
    for (int tt = 0; tt < 4; ++tt) accO[tt] = v8f_zero();
    float mrow[8], lrow[8];
    #pragma unroll
    for (int r = 0; r < 8; ++r) { mrow[r] = -1e30f; lrow[r] = 0.0f; }

    auto loadKV = [&](int bufi, int kb) {
        int keyBase = kb * 32;
        #pragma unroll
        for (int idx = t; idx < 256; idx += 128) {      // K: 32 keys x 64 d
            int row = idx >> 3, c8 = idx & 7;
            stage16(&Ks[bufi][row][c8 * 8],
                    &Kc[boff + (size_t)(keyBase + row) * HD + c8 * 8]);
        }
        #pragma unroll
        for (int idx = t; idx < 256; idx += 128) {      // V^T: 64 d x 32 keys
            int row = idx >> 2, c8 = idx & 3;
            stage16(&Vs[bufi][row][c8 * 8],
                    &VT[boff + (size_t)row * SEQ + keyBase + c8 * 8]);
        }
    };

    loadKV(0, 0);
    int buf = 0;
    for (int kb = 0; kb < SEQ / 32; ++kb) {
        stage_wait();
        __syncthreads();
        if (kb + 1 < SEQ / 32) loadKV(buf ^ 1, kb + 1);

        // S = Q K^T  (two 16x16 key tiles, K-dim = 64 via two WMMAs each)
        v8f s[2];
        #pragma unroll
        for (int j = 0; j < 2; ++j) {
            v8f sj = v8f_zero();
            #pragma unroll
            for (int c = 0; c < 2; ++c) {
                Frag16 bk;
                int key = j * 16 + l16;
                bk.u[0] = *(const uint4*)&Ks[buf][key][c * 32 + half * 8];
                bk.u[1] = *(const uint4*)&Ks[buf][key][c * 32 + 16 + half * 8];
                sj = __builtin_amdgcn_wmma_f32_16x16x32_bf16(
                    false, aq[c].v, false, bk.v, (short)0, sj, false, false);
            }
            s[j] = sj;
        }

        // Online softmax: element r of the C-fragment = row (half*8+r), so the
        // row reduction is shfl_xor over the 16-lane half -> exact broadcast.
        float corr[8];
        #pragma unroll
        for (int r = 0; r < 8; ++r) {
            float s0 = s[0][r] * SCALE_F, s1 = s[1][r] * SCALE_F;
            s[0][r] = s0; s[1][r] = s1;
            float mx = fmaxf(s0, s1);
            #pragma unroll
            for (int off = 1; off < 16; off <<= 1) mx = fmaxf(mx, __shfl_xor(mx, off, 32));
            float mnew = fmaxf(mrow[r], mx);
            corr[r] = exp2f((mrow[r] - mnew) * L2E);
            mrow[r] = mnew;
        }
        #pragma unroll
        for (int r = 0; r < 8; ++r) {
            float p0 = exp2f((s[0][r] - mrow[r]) * L2E);
            float p1 = exp2f((s[1][r] - mrow[r]) * L2E);
            s[0][r] = p0; s[1][r] = p1;
            float rs = p0 + p1;
            #pragma unroll
            for (int off = 1; off < 16; off <<= 1) rs += __shfl_xor(rs, off, 32);
            lrow[r] = lrow[r] * corr[r] + rs;
            #pragma unroll
            for (int tt = 0; tt < 4; ++tt) accO[tt][r] *= corr[r];
        }

        // Transpose P (C layout -> A layout) through wave-private LDS.
        #pragma unroll
        for (int r = 0; r < 8; ++r) {
            int row = half * 8 + r;
            Ps[wave][row][l16]      = f2bf(s[0][r]);
            Ps[wave][row][16 + l16] = f2bf(s[1][r]);
        }
        Frag16 ap;
        ap.u[0] = *(const uint4*)&Ps[wave][l16][half * 8];
        ap.u[1] = *(const uint4*)&Ps[wave][l16][16 + half * 8];

        // accO += P @ V  (four 16-wide d tiles)
        #pragma unroll
        for (int tt = 0; tt < 4; ++tt) {
            Frag16 bv;
            int d = tt * 16 + l16;
            bv.u[0] = *(const uint4*)&Vs[buf][d][half * 8];
            bv.u[1] = *(const uint4*)&Vs[buf][d][16 + half * 8];
            accO[tt] = __builtin_amdgcn_wmma_f32_16x16x32_bf16(
                false, ap.v, false, bv.v, (short)0, accO[tt], false, false);
        }
        buf ^= 1;
    }

    // Epilogue: ctx[b*SEQ + row][h*64 + d] = accO / l
    #pragma unroll
    for (int tt = 0; tt < 4; ++tt) {
        int d = tt * 16 + l16;
        #pragma unroll
        for (int r = 0; r < 8; ++r) {
            int qrow = qBase + wave * 16 + half * 8 + r;
            ctx[(size_t)(b * SEQ + qrow) * DIM + h * HD + d] = f2bf(accO[tt][r] / lrow[r]);
        }
    }
}

// ---------------------------------------------------------------------------
// GEMM2: out = ctx @ W_out + b_out   (fp32 output). M=8192, N=768, K=768.
// ---------------------------------------------------------------------------
__global__ __launch_bounds__(256)
void gemm_out(const ushort_t* __restrict__ Ab,    // ctx [8192][768] bf16
              const ushort_t* __restrict__ Bt,    // [768][768] bf16 (W_out^T)
              const float*    __restrict__ bias,  // [768]
              float* __restrict__ Out)            // [8192][768] fp32
{
    __shared__ __align__(16) ushort_t As[2][128][32];
    __shared__ __align__(16) ushort_t Bs[2][128][32];

    const int mBase = blockIdx.x * 128;
    const int nBase = blockIdx.y * 128;
    const int t    = threadIdx.x;
    const int wave = t >> 5;
    const int lane = t & 31;
    const int half = lane >> 4;
    const int l16  = lane & 15;
    const int wm   = (wave & 3) * 32;
    const int wn   = (wave >> 2) * 64;

    v8f acc[2][4];
    #pragma unroll
    for (int i = 0; i < 2; ++i)
        #pragma unroll
        for (int j = 0; j < 4; ++j) acc[i][j] = v8f_zero();

    auto loadTiles = [&](int buf, int k0) {
        #pragma unroll
        for (int idx = t; idx < 512; idx += 256) {
            int row = idx >> 2, c8 = idx & 3;
            stage16(&As[buf][row][c8 * 8], &Ab[(size_t)(mBase + row) * DIM + k0 + c8 * 8]);
            stage16(&Bs[buf][row][c8 * 8], &Bt[(size_t)(nBase + row) * DIM + k0 + c8 * 8]);
        }
    };

    loadTiles(0, 0);
    int buf = 0;
    for (int k0 = 0; k0 < DIM; k0 += 32) {
        stage_wait();
        __syncthreads();
        if (k0 + 32 < DIM) loadTiles(buf ^ 1, k0 + 32);
        Frag16 a[2], b[4];
        #pragma unroll
        for (int i = 0; i < 2; ++i) {
            int m = wm + i * 16 + l16;
            a[i].u[0] = *(const uint4*)&As[buf][m][half * 8];
            a[i].u[1] = *(const uint4*)&As[buf][m][16 + half * 8];
        }
        #pragma unroll
        for (int j = 0; j < 4; ++j) {
            int n = wn + j * 16 + l16;
            b[j].u[0] = *(const uint4*)&Bs[buf][n][half * 8];
            b[j].u[1] = *(const uint4*)&Bs[buf][n][16 + half * 8];
        }
        #pragma unroll
        for (int i = 0; i < 2; ++i)
            #pragma unroll
            for (int j = 0; j < 4; ++j)
                acc[i][j] = __builtin_amdgcn_wmma_f32_16x16x32_bf16(
                    false, a[i].v, false, b[j].v, (short)0, acc[i][j], false, false);
        buf ^= 1;
    }

    #pragma unroll
    for (int j = 0; j < 4; ++j) {
        int c = nBase + wn + j * 16 + l16;
        float bv = bias[c];
        #pragma unroll
        for (int i = 0; i < 2; ++i) {
            #pragma unroll
            for (int r = 0; r < 8; ++r) {
                int m = mBase + wm + i * 16 + half * 8 + r;
                Out[(size_t)m * DIM + c] = acc[i][j][r] + bv;
            }
        }
    }
}

// ---------------------------------------------------------------------------
// Host-side launcher
// ---------------------------------------------------------------------------
extern "C" void kernel_launch(void* const* d_in, const int* in_sizes, int n_in,
                              void* d_out, int out_size, void* d_ws, size_t ws_size,
                              hipStream_t stream) {
    (void)in_sizes; (void)n_in; (void)out_size; (void)ws_size;
    const float* x     = (const float*)d_in[0];
    const float* W_qkv = (const float*)d_in[1];
    const float* b_qkv = (const float*)d_in[2];
    const float* W_out = (const float*)d_in[3];
    const float* b_out = (const float*)d_in[4];
    float* out = (float*)d_out;

    char* ws = (char*)d_ws;
    size_t off = 0;
    auto carve = [&](size_t bytes) -> void* {
        void* p = ws + off;
        off += (bytes + 255) & ~(size_t)255;
        return p;
    };
    ushort_t* xb    = (ushort_t*)carve((size_t)M_TOT * DIM * 2);
    ushort_t* wqkvT = (ushort_t*)carve((size_t)C3 * DIM * 2);
    ushort_t* woutT = (ushort_t*)carve((size_t)DIM * DIM * 2);
    ushort_t* Qb    = (ushort_t*)carve((size_t)BHTOT * SEQ * HD * 2);
    ushort_t* Kb    = (ushort_t*)carve((size_t)BHTOT * SEQ * HD * 2);
    ushort_t* VTb   = (ushort_t*)carve((size_t)BHTOT * SEQ * HD * 2);
    ushort_t* ctx   = (ushort_t*)carve((size_t)M_TOT * DIM * 2);

    pack_bf16          <<<512, 256, 0, stream>>>(x, xb, M_TOT * DIM);
    pack_transpose_bf16<<<512, 256, 0, stream>>>(W_qkv, wqkvT, DIM, C3);
    pack_transpose_bf16<<<256, 256, 0, stream>>>(W_out, woutT, DIM, DIM);

    gemm_qkv  <<<dim3(M_TOT / 128, C3 / 128),  256, 0, stream>>>(xb, wqkvT, b_qkv, Qb, Kb, VTb);
    flash_attn<<<dim3(BHTOT, SEQ / 64),        128, 0, stream>>>(Qb, Kb, VTb, ctx);
    gemm_out  <<<dim3(M_TOT / 128, DIM / 128), 256, 0, stream>>>(ctx, woutT, b_out, out);
}
